// BoundaryIBSDE_6786048327947
// MI455X (gfx1250) — matrix operand order repaired and checked by reading, not compiled, not run on hardware
//
#include <hip/hip_runtime.h>

// BSDE with reflecting/absorbing boundaries on MI455X (gfx1250, wave32).
// One persistent kernel: 256 WGs x 256 threads; lane == path; 64 time steps
// with per-step 64x64 layer done via v_wmma_f32_16x16x32_bf16.

#define B_TOT   65536
#define N_STEPS 64
#define DT_C    (1.0f / 64.0f)
#define SIG_C   1.4142135623730951f
#define ROUT_C  1.0f
#define RHIT_C  0.3f

typedef __attribute__((ext_vector_type(16))) __bf16        v16bf;
typedef __attribute__((ext_vector_type(8)))  float         v8f;
typedef __attribute__((ext_vector_type(8)))  unsigned int  v8u;

__device__ __forceinline__ unsigned short f2bf(float f) {
  unsigned u = __builtin_bit_cast(unsigned, f);
  u += 0x7FFFu + ((u >> 16) & 1u);          // round-to-nearest-even
  return (unsigned short)(u >> 16);
}
__device__ __forceinline__ unsigned packbf(float a, float b) {
  return (unsigned)f2bf(a) | ((unsigned)f2bf(b) << 16);
}
__device__ __forceinline__ float bf2f(unsigned short h) {
  return __builtin_bit_cast(float, ((unsigned)h) << 16);
}
__device__ __forceinline__ float fast_tanh(float x) {
#if __has_builtin(__builtin_amdgcn_tanhf)
  return __builtin_amdgcn_tanhf(x);          // v_tanh_f32 on gfx1250
#elif __has_builtin(__builtin_amdgcn_tanh_f32)
  return __builtin_amdgcn_tanh_f32(x);
#else
  float e = __expf(2.0f * x);                // v_exp_f32 path
  return 1.0f - 2.0f / (e + 1.0f);
#endif
}

__global__ __launch_bounds__(256) void bsde_fused_kernel(
    const float* __restrict__ x0g, const float* __restrict__ dBt,
    const float* __restrict__ pW1, const float* __restrict__ pb1,
    const float* __restrict__ pW2, const float* __restrict__ pb2,
    const float* __restrict__ pW3, const float* __restrict__ pb3,
    const float* __restrict__ gW1, const float* __restrict__ gb1,
    const float* __restrict__ gW2, const float* __restrict__ gb2,
    const float* __restrict__ gW3, const float* __restrict__ gb3,
    float* __restrict__ out)
{
  // LDS: W2^T in bf16 (B fragments), small f32 weights, 256x64 bf16 h-slab.
  __shared__ __align__(16) unsigned short sW2t[64 * 64];   // [n][k] bf16, 8KB
  __shared__ float sW1[3 * 64];
  __shared__ float sb1[64];
  __shared__ float sb2[64];
  __shared__ float sW3[64 * 3];
  __shared__ float sb3[4];
  __shared__ __align__(16) unsigned short sH[256 * 64];    // 32KB, reused

  const int tid  = threadIdx.x;
  const int wave = tid >> 5;
  const int lane = tid & 31;
  const int ln16 = lane & 15;
  const int lhi  = lane >> 4;                  // 0 or 1 (lane half)
  const int row  = blockIdx.x * 256 + tid;     // global path id

  // ---------------- per-path state (registers for all 64 steps) -----------
  float x0 = x0g[row * 3 + 0];
  float x1 = x0g[row * 3 + 1];
  float x2 = x0g[row * 3 + 2];
  float p, prel = 0.0f;
  bool  run = true;

  // ---------------- p0 = fnn(x0; pW*)  (one-time, per-lane f32) -----------
  {
    float* spw2 = (float*)sH;                  // reuse 32KB slab for 16KB f32 W2
    for (int i = tid; i < 4096; i += 256) spw2[i] = pW2[i];
    if (tid < 192) sW1[tid] = pW1[tid];
    if (tid < 64)  { sb1[tid] = pb1[tid]; sb2[tid] = pb2[tid]; sW3[tid] = pW3[tid]; }
    if (tid == 0)  sb3[0] = pb3[0];
    __syncthreads();

    float h1[64];
#pragma unroll
    for (int k = 0; k < 64; ++k)
      h1[k] = fast_tanh(sb1[k] + x0 * sW1[k] + x1 * sW1[64 + k] + x2 * sW1[128 + k]);
    float acc = sb3[0];
    for (int j = 0; j < 64; ++j) {
      float a = sb2[j];
#pragma unroll
      for (int i = 0; i < 64; ++i) a += h1[i] * spw2[i * 64 + j];
      acc += fast_tanh(a) * sW3[j];
    }
    p = acc;
    __syncthreads();
  }

  // ---------------- time loop ---------------------------------------------
  for (int n = 0; n < N_STEPS; ++n) {
    // ---- cooperative load of this step's weights into LDS ----
    if (tid < 192) sW1[tid] = gW1[n * 192 + tid];
    if (tid < 64)  { sb1[tid] = gb1[n * 64 + tid]; sb2[tid] = gb2[n * 64 + tid]; }
    if (tid < 192) sW3[tid] = gW3[n * 192 + tid];
    if (tid < 3)   sb3[tid] = gb3[n * 3 + tid];
    {
      // W2[i][j] -> sW2t[j][i] as bf16 pairs (i even), dword stores.
      const float* src = gW2 + (size_t)n * 4096;
      int j = tid & 63, ic = tid >> 6;         // 4 threads per column j
      unsigned* dst = (unsigned*)sW2t;
#pragma unroll
      for (int u = 0; u < 8; ++u) {
        int i = ic * 16 + 2 * u;
        dst[j * 32 + (i >> 1)] = packbf(src[i * 64 + j], src[(i + 1) * 64 + j]);
      }
    }
    __syncthreads();

    // ---- layer 1 (per-lane): h1 = tanh(x @ W1 + b1) -> sH bf16 [row][k] ---
    {
      unsigned* hrow = ((unsigned*)sH) + tid * 32;
#pragma unroll
      for (int k = 0; k < 64; k += 2) {
        float a = fast_tanh(sb1[k]     + x0 * sW1[k]       + x1 * sW1[64 + k]     + x2 * sW1[128 + k]);
        float b = fast_tanh(sb1[k + 1] + x0 * sW1[k + 1]   + x1 * sW1[64 + k + 1] + x2 * sW1[128 + k + 1]);
        hrow[k >> 1] = packbf(a, b);
      }
    }
    __syncthreads();

    // ---- layer 2 via WMMA: h2 = tanh(h1 @ W2 + b2) ----
    // B fragments (contiguous K order, same as A): lane -> N = 16c+ln16,
    // K = 32q + 16*lhi + 2v + {0,1}; sW2t is [N][K] so K pairs are adjacent.
    v8u bfr[4][2];
    {
      const unsigned* w2t = (const unsigned*)sW2t;
#pragma unroll
      for (int c = 0; c < 4; ++c) {
        int jn = 16 * c + ln16;
#pragma unroll
        for (int q = 0; q < 2; ++q) {
          int base = jn * 32 + 16 * q + 8 * lhi;
#pragma unroll
          for (int v = 0; v < 8; ++v) bfr[c][q][v] = w2t[base + v];
        }
      }
    }
    const unsigned* hsl = (const unsigned*)sH;
#pragma unroll
    for (int t = 0; t < 2; ++t) {              // two 16-row tiles per wave
      // A fragments (16-bit A 16x32): M = ln16, K-pair at
      // 2v + 16*lhi, plus 32 for the second K chunk.
      v8u av0, av1;
      int rowb  = wave * 32 + 16 * t + ln16;
      int khalf = 16 * lhi;
#pragma unroll
      for (int v = 0; v < 8; ++v) {
        int kv = 2 * v + khalf;
        av0[v] = hsl[rowb * 32 + (kv >> 1)];
        av1[v] = hsl[rowb * 32 + ((kv + 32) >> 1)];
      }
      v16bf a0 = __builtin_bit_cast(v16bf, av0);
      v16bf a1 = __builtin_bit_cast(v16bf, av1);
#pragma unroll
      for (int c = 0; c < 4; ++c) {
        float bias = sb2[16 * c + ln16];       // C layout: N = ln16 for all VGPRs
        v8f acc;
#pragma unroll
        for (int r = 0; r < 8; ++r) acc[r] = bias;
        acc = __builtin_amdgcn_wmma_f32_16x16x32_bf16(
                false, a0, false, __builtin_bit_cast(v16bf, bfr[c][0]),
                (short)0, acc, false, false);
        acc = __builtin_amdgcn_wmma_f32_16x16x32_bf16(
                false, a1, false, __builtin_bit_cast(v16bf, bfr[c][1]),
                (short)0, acc, false, false);
        // C/D layout: lane L holds N=ln16, M = r + 8*lhi; tanh + bf16 store.
#pragma unroll
        for (int r = 0; r < 8; ++r) {
          int row2 = wave * 32 + 16 * t + r + 8 * lhi;
          sH[row2 * 64 + 16 * c + ln16] = f2bf(fast_tanh(acc[r]));
        }
      }
    }
    __syncthreads();

    // ---- layer 3 (per-lane): grad_p = h2 @ W3 + b3 ----
    float g0 = sb3[0], g1 = sb3[1], g2 = sb3[2];
    {
      const unsigned* hrow = ((const unsigned*)sH) + tid * 32;
#pragma unroll
      for (int k2 = 0; k2 < 32; ++k2) {
        unsigned d = hrow[k2];
        float ha = bf2f((unsigned short)(d & 0xFFFFu));
        float hb = bf2f((unsigned short)(d >> 16));
        int k = 2 * k2;
        g0 += ha * sW3[k * 3 + 0] + hb * sW3[k * 3 + 3];
        g1 += ha * sW3[k * 3 + 1] + hb * sW3[k * 3 + 4];
        g2 += ha * sW3[k * 3 + 2] + hb * sW3[k * 3 + 5];
      }
    }

    // ---- SDE + BSDE update (per-lane, f32) ----
    float f = p + 0.5f * (g0 * g0 + g1 * g1 + g2 * g2);
    const float* db = dBt + ((size_t)n * B_TOT + row) * 3;
    float dB0 = db[0], dB1 = db[1], dB2 = db[2];
    if (run) {
      x0 += -0.5f * x0 * DT_C + SIG_C * dB0;
      x1 += -0.5f * x1 * DT_C + SIG_C * dB1;
      x2 += -0.5f * x2 * DT_C + SIG_C * dB2;
    }
    float nrm = sqrtf(x0 * x0 + x1 * x1 + x2 * x2);
    if (nrm > ROUT_C) {                         // mirror reflect at |x| = R_OUT
      float s = (2.0f * ROUT_C - nrm) / nrm;
      x0 *= s; x1 *= s; x2 *= s;
    }
    float nrm2 = x0 * x0 + x1 * x1 + x2 * x2;
    bool hit = (nrm2 <= RHIT_C * RHIT_C);
    float dp = -f * DT_C + SIG_C * (g0 * dB0 + g1 * dB1 + g2 * dB2);
    if (run) {
      p += dp;
      if (hit) { prel = nrm2; run = false; }    // pc(xt) = |xt|^2
    }
    __syncthreads();                            // sH/weights reuse next step
  }

  if (run) prel = x0 * x0 + x1 * x1 + x2 * x2;
  out[row]         = p;                         // output 0: p      [B,1]
  out[B_TOT + row] = prel;                      // output 1: p_rel  [B,1]
}

extern "C" void kernel_launch(void* const* d_in, const int* in_sizes, int n_in,
                              void* d_out, int out_size, void* d_ws, size_t ws_size,
                              hipStream_t stream) {
  (void)in_sizes; (void)n_in; (void)out_size; (void)d_ws; (void)ws_size;
  const float* x0  = (const float*)d_in[0];
  const float* dBt = (const float*)d_in[1];
  const float* pW1 = (const float*)d_in[2];
  const float* pb1 = (const float*)d_in[3];
  const float* pW2 = (const float*)d_in[4];
  const float* pb2 = (const float*)d_in[5];
  const float* pW3 = (const float*)d_in[6];
  const float* pb3 = (const float*)d_in[7];
  const float* gW1 = (const float*)d_in[8];
  const float* gb1 = (const float*)d_in[9];
  const float* gW2 = (const float*)d_in[10];
  const float* gb2 = (const float*)d_in[11];
  const float* gW3 = (const float*)d_in[12];
  const float* gb3 = (const float*)d_in[13];
  float* out = (float*)d_out;

  dim3 grid(B_TOT / 256), block(256);
  bsde_fused_kernel<<<grid, block, 0, stream>>>(
      x0, dBt, pW1, pb1, pW2, pb2, pW3, pb3,
      gW1, gb1, gW2, gb2, gW3, gb3, out);
}
